// MUTAN_34935263986216
// MI455X (gfx1250) — matrix operand order repaired and checked by reading, not compiled, not run on hardware
//
#include <hip/hip_runtime.h>

typedef __attribute__((ext_vector_type(16))) _Float16 v16h;
typedef __attribute__((ext_vector_type(8)))  _Float16 v8h;
typedef __attribute__((ext_vector_type(2)))  __fp16   v2hp;   // matches cvt_pkrtz return
typedef __attribute__((ext_vector_type(8)))  float    v8f;

#define B_   4096
#define D_   1024
#define Z_   1024
#define R_   16
#define BM   64        // batch rows per workgroup (2 wave rows x 2 M-subtiles x 16)
#define BN   128       // z cols per workgroup   (4 wave cols x 2 N-subtiles x 16)
#define BK   32        // K step (matches wmma 16x16x32)
#define BKP  40        // padded K stride in halves (80 B: 16B-aligned, bank-spread)
#define KT_  (D_ / BK)

// -------------------------------------------------------------------------
// Kernel 1: fused low-rank bilinear  z[b,n] = relu( sum_r (hd@U[r]) * (hp@V[r]) )
// f32 -> f16 packed conversion on the fly, f32 accumulation in WMMA.
// Each wave computes a 32x32 output macro-tile (2x2 WMMA subtiles) for both
// the u and v GEMMs; A/B fragments are reused across the 2x2 grid.
// -------------------------------------------------------------------------
__global__ __launch_bounds__(256) void mutan_bilinear_wmma(
    const float* __restrict__ hd, const float* __restrict__ hp,
    const float* __restrict__ U,  const float* __restrict__ V,
    float* __restrict__ zbuf)
{
    // A tiles row-major [m][k] (padded), B tiles column-major [n][k] (padded)
    __shared__ _Float16 sAu[BM * BKP];
    __shared__ _Float16 sAv[BM * BKP];
    __shared__ _Float16 sBu[BN * BKP];
    __shared__ _Float16 sBv[BN * BKP];

    const int tid  = threadIdx.x;
    const int wave = tid >> 5;
    const int lane = tid & 31;
    const int wm   = wave & 1;      // 2 wave rows,  2 M-subtiles each
    const int wn   = wave >> 1;     // 4 wave cols,  2 N-subtiles each
    const int m0   = blockIdx.y * BM;
    const int n0   = blockIdx.x * BN;

    const int lrow  = lane & 15;    // row (A) / col (B) / N (C,D) within 16
    const int lhalf = lane >> 4;    // lane half selects K group / M group

    // Loop-invariant 32-bit per-thread element offsets (enables saddr+voffset loads)
    unsigned aofs[4];               // A: 1024 pairs per matrix -> 4 iters
#pragma unroll
    for (int i = 0; i < 4; ++i) {
        const int p = tid + i * 256;
        aofs[i] = (unsigned)((p >> 4) * D_ + (p & 15) * 2);   // m*D + kp
    }
    unsigned bofs[8];               // B: 2048 (n,kpair) items per matrix -> 8 iters
#pragma unroll
    for (int i = 0; i < 8; ++i) {
        const int q = tid + i * 256;
        bofs[i] = (unsigned)(((q >> 7) * 2) * Z_ + (q & 127)); // kp*Z + n
    }
    // Matching LDS element offsets
    unsigned aldst[4];
#pragma unroll
    for (int i = 0; i < 4; ++i) {
        const int p = tid + i * 256;
        aldst[i] = (unsigned)((p >> 4) * BKP + (p & 15) * 2);
    }
    unsigned bldst[8];
#pragma unroll
    for (int i = 0; i < 8; ++i) {
        const int q = tid + i * 256;
        bldst[i] = (unsigned)((q & 127) * BKP + (q >> 7) * 2);
    }

    v8f zacc[2][2] = {};

    for (int r = 0; r < R_; ++r) {
        const float* __restrict__ Ur = U + (size_t)r * D_ * Z_ + n0;
        const float* __restrict__ Vr = V + (size_t)r * D_ * Z_ + n0;
        v8f uacc[2][2] = {};
        v8f vacc[2][2] = {};

        for (int kt = 0; kt < KT_; ++kt) {
            const int k0 = kt * BK;
            const float* __restrict__ Ad = hd + (size_t)m0 * D_ + k0;   // uniform bases
            const float* __restrict__ Ap = hp + (size_t)m0 * D_ + k0;
            const float* __restrict__ Bu = Ur + (size_t)k0 * Z_;
            const float* __restrict__ Bv = Vr + (size_t)k0 * Z_;

            // ---- stage A tiles (BM x BK): float2 loads + packed f16 stores ----
#pragma unroll
            for (int i = 0; i < 4; ++i) {
                const float2 du = *(const float2*)(&Ad[aofs[i]]);
                const float2 dp = *(const float2*)(&Ap[aofs[i]]);
                *(v2hp*)(&sAu[aldst[i]]) = __builtin_amdgcn_cvt_pkrtz(du.x, du.y);
                *(v2hp*)(&sAv[aldst[i]]) = __builtin_amdgcn_cvt_pkrtz(dp.x, dp.y);
            }
            // ---- stage B tiles (BK x BN) -> column-major [n][k], packed pairs ----
#pragma unroll
            for (int i = 0; i < 8; ++i) {
                const float u0 = Bu[bofs[i]];
                const float u1 = Bu[bofs[i] + Z_];
                const float w0 = Bv[bofs[i]];
                const float w1 = Bv[bofs[i] + Z_];
                *(v2hp*)(&sBu[bldst[i]]) = __builtin_amdgcn_cvt_pkrtz(u0, u1);
                *(v2hp*)(&sBv[bldst[i]]) = __builtin_amdgcn_cvt_pkrtz(w0, w1);
            }
            // hint: pull next K tile of U/V toward the caches
            if (kt + 1 < KT_) {
                __builtin_prefetch(&Bu[(size_t)BK * Z_ + bofs[0]], 0, 1);
                __builtin_prefetch(&Bv[(size_t)BK * Z_ + bofs[0]], 0, 1);
            }
            __syncthreads();

            // ---- A fragments (2 M-subtiles): ISA 16-bit A 16x32 layout ----
            // lane half h holds K in [8h,8h+8) (VGPR0-3) and [16+8h,16+8h+8) (VGPR4-7)
            v16h au[2], av[2];
#pragma unroll
            for (int mi = 0; mi < 2; ++mi) {
                const int mrow = ((wm * 2 + mi) * 16 + lrow) * BKP;
                *((v8h*)&au[mi] + 0) = *(const v8h*)(&sAu[mrow + lhalf * 8]);
                *((v8h*)&au[mi] + 1) = *(const v8h*)(&sAu[mrow + 16 + lhalf * 8]);
                *((v8h*)&av[mi] + 0) = *(const v8h*)(&sAv[mrow + lhalf * 8]);
                *((v8h*)&av[mi] + 1) = *(const v8h*)(&sAv[mrow + 16 + lhalf * 8]);
            }
            // ---- B fragments (2 N-subtiles): lane half h holds K in [8h,8h+8) and [16+8h,16+8h+8) ----
            v16h bu[2], bv[2];
#pragma unroll
            for (int ni = 0; ni < 2; ++ni) {
                const int ncol = ((wn * 2 + ni) * 16 + lrow) * BKP;
                *((v8h*)&bu[ni] + 0) = *(const v8h*)(&sBu[ncol + lhalf * 8]);
                *((v8h*)&bu[ni] + 1) = *(const v8h*)(&sBu[ncol + 16 + lhalf * 8]);
                *((v8h*)&bv[ni] + 0) = *(const v8h*)(&sBv[ncol + lhalf * 8]);
                *((v8h*)&bv[ni] + 1) = *(const v8h*)(&sBv[ncol + 16 + lhalf * 8]);
            }

            // ---- 2x2 WMMA grid for u and for v ----
#pragma unroll
            for (int mi = 0; mi < 2; ++mi) {
#pragma unroll
                for (int ni = 0; ni < 2; ++ni) {
                    uacc[mi][ni] = __builtin_amdgcn_wmma_f32_16x16x32_f16(
                        false, au[mi], false, bu[ni], (short)0, uacc[mi][ni], false, false);
                    vacc[mi][ni] = __builtin_amdgcn_wmma_f32_16x16x32_f16(
                        false, av[mi], false, bv[ni], (short)0, vacc[mi][ni], false, false);
                }
            }

            __syncthreads();  // protect LDS before next restage
        }

        // rank contraction: z += u * v (elementwise on the 16x16 tiles)
#pragma unroll
        for (int mi = 0; mi < 2; ++mi)
#pragma unroll
            for (int ni = 0; ni < 2; ++ni)
#pragma unroll
                for (int i = 0; i < 8; ++i)
                    zacc[mi][ni][i] += uacc[mi][ni][i] * vacc[mi][ni][i];
    }

    // ReLU + store pre-LayerNorm z to workspace.
    // C/D layout: lanes 0-15: N=lane, M=i ; lanes 16-31: N=lane-16, M=8+i
#pragma unroll
    for (int mi = 0; mi < 2; ++mi) {
#pragma unroll
        for (int ni = 0; ni < 2; ++ni) {
            const int col = n0 + (wn * 2 + ni) * 16 + lrow;
#pragma unroll
            for (int i = 0; i < 8; ++i) {
                const int row = m0 + (wm * 2 + mi) * 16 + i + 8 * lhalf;
                const float zv = zacc[mi][ni][i] > 0.f ? zacc[mi][ni][i] : 0.f;
                zbuf[(size_t)row * Z_ + col] = zv;
            }
        }
    }
}

// -------------------------------------------------------------------------
// Kernel 2: row-wise LayerNorm over z (1024), gamma/beta affine.
// One block (256 threads) per batch row; 4 elements per thread.
// -------------------------------------------------------------------------
__global__ __launch_bounds__(256) void mutan_layernorm(
    const float* __restrict__ zbuf, const float* __restrict__ gamma,
    const float* __restrict__ beta, float* __restrict__ out)
{
    __shared__ float ssum[256];
    __shared__ float ssq[256];

    const int row = blockIdx.x;
    const int tid = threadIdx.x;

    const float4 xv = *(const float4*)(&zbuf[(size_t)row * Z_ + tid * 4]);
    float s = xv.x + xv.y + xv.z + xv.w;
    float q = xv.x * xv.x + xv.y * xv.y + xv.z * xv.z + xv.w * xv.w;
    ssum[tid] = s;
    ssq[tid]  = q;
    __syncthreads();

#pragma unroll
    for (int off = 128; off > 0; off >>= 1) {
        if (tid < off) {
            ssum[tid] += ssum[tid + off];
            ssq[tid]  += ssq[tid + off];
        }
        __syncthreads();
    }

    const float mean = ssum[0] * (1.0f / Z_);
    const float var  = ssq[0] * (1.0f / Z_) - mean * mean;
    const float inv  = rsqrtf(var + 1e-5f);

    const float4 g = *(const float4*)(&gamma[tid * 4]);
    const float4 b = *(const float4*)(&beta[tid * 4]);
    float4 o;
    o.x = (xv.x - mean) * inv * g.x + b.x;
    o.y = (xv.y - mean) * inv * g.y + b.y;
    o.z = (xv.z - mean) * inv * g.z + b.z;
    o.w = (xv.w - mean) * inv * g.w + b.w;
    *(float4*)(&out[(size_t)row * Z_ + tid * 4]) = o;
}

// -------------------------------------------------------------------------
extern "C" void kernel_launch(void* const* d_in, const int* in_sizes, int n_in,
                              void* d_out, int out_size, void* d_ws, size_t ws_size,
                              hipStream_t stream)
{
    const float* hd    = (const float*)d_in[0];
    const float* hp    = (const float*)d_in[1];
    const float* U     = (const float*)d_in[2];
    const float* V     = (const float*)d_in[3];
    const float* gamma = (const float*)d_in[4];
    const float* beta  = (const float*)d_in[5];
    float* out  = (float*)d_out;
    float* zbuf = (float*)d_ws;   // needs B_*Z_*4 = 16 MiB scratch

    dim3 grid1(Z_ / BN, B_ / BM);    // 8 x 64 workgroups
    mutan_bilinear_wmma<<<grid1, 256, 0, stream>>>(hd, hp, U, V, zbuf);
    mutan_layernorm<<<B_, 256, 0, stream>>>(zbuf, gamma, beta, out);
}
